// LSTM_14482629722403
// MI455X (gfx1250) — matrix operand order repaired and checked by reading, not compiled, not run on hardware
//
#include <hip/hip_runtime.h>
#include <hip/hip_bf16.h>
#include <stdint.h>
#include <stddef.h>

// ---------------------------------------------------------------------------
// Types for CDNA5 WMMA (gfx1250, wave32)
// ---------------------------------------------------------------------------
typedef __bf16 bf16;
typedef __attribute__((ext_vector_type(16))) __bf16 v16bf;
typedef __attribute__((ext_vector_type(8)))  __bf16 v8bf;
typedef __attribute__((ext_vector_type(8)))  float  v8f;

#define LN_EPS 1e-5f

__device__ __forceinline__ bf16 f2bf(float f) {
  union { float f; uint32_t u; } v; v.f = f;
  uint32_t r = (v.u + 0x7FFFu + ((v.u >> 16) & 1u)) >> 16;  // RNE
  uint16_t s = (uint16_t)r;
  bf16 o; __builtin_memcpy(&o, &s, 2);
  return o;
}

__device__ __forceinline__ v16bf mk16(v8bf lo, v8bf hi) {
  return __builtin_shufflevector(lo, hi, 0,1,2,3,4,5,6,7,8,9,10,11,12,13,14,15);
}

// ---------------------------------------------------------------------------
// Swizzle fp32 weights W[K,N] (row-major) into WMMA B-fragment layout:
//   Wf[((nt*KT + kt)*32 + lane)*16 + e] = bf16( W[kt*32 + (lane>>4)*8 + (e&7) + (e>>3)*16][nt*16 + (lane&15)] )
// so a wave's B fragment for tile (kt,nt) is 32 contiguous bytes per lane,
// fully coalesced (1KB per wave), and L2-resident for the whole scan.
// ---------------------------------------------------------------------------
__global__ __launch_bounds__(256)
void swizzle_wf(const float* __restrict__ W, bf16* __restrict__ Wf, int K, int N) {
  int o = blockIdx.x * 256 + threadIdx.x;
  if (o >= K * N) return;
  int e  = o & 15;
  int L  = (o >> 4) & 31;
  int t  = o >> 9;            // nt*KT + kt
  int KT = K >> 5;
  int kt = t % KT;
  int nt = t / KT;
  int k  = kt * 32 + ((L >> 4) << 3) + (e & 7) + ((e >> 3) << 4);
  int n  = nt * 16 + (L & 15);
  Wf[o] = f2bf(W[(size_t)k * (size_t)N + (size_t)n]);
}

// zero-init h (bf16) and c (fp32) state
__global__ __launch_bounds__(256) void init_state(float* __restrict__ c,
                                                  bf16* __restrict__ h, int n) {
  int i = blockIdx.x * 256 + threadIdx.x;
  if (i < n) { c[i] = 0.0f; h[i] = f2bf(0.0f); }
}

// ---------------------------------------------------------------------------
// Recurrent-step GEMM (latency critical, on the T=512 serial chain):
//   pre[64,2048] = h[64,512] @ Wh[512,2048] + bh
// Register-direct WMMA: no LDS, no barriers. One wave per 16x16 C tile,
// 512 waves = 64 blocks x 8 waves. A frags: two aligned b128 loads from
// row-major bf16 h (64KB, L2/WGP$-hot). B frags: contiguous 32B/lane from
// pre-swizzled Whf. Fully unrolled 16-deep WMMA chain.
// ---------------------------------------------------------------------------
__global__ __launch_bounds__(256)
void wmma_gemm_step(const bf16* __restrict__ A,    // [64,512] row-major
                    const bf16* __restrict__ Bf,   // swizzled [nt][kt][lane][16]
                    const float* __restrict__ bias,
                    float* __restrict__ Craw) {    // [64,2048]
  constexpr int K = 512, N = 2048, KT = K / 32;
  const int tid  = threadIdx.x;
  const int lane = tid & 31;
  const int gw   = blockIdx.x * 8 + (tid >> 5);  // 0..511
  const int mt   = gw & 3;                        // 4 M tiles
  const int nt   = gw >> 2;                       // 128 N tiles
  const int rc    = lane & 15;
  const int khalf = (lane >> 4) << 3;

  const bf16* arow  = A + (size_t)(mt * 16 + rc) * K;
  const bf16* bbase = Bf + ((size_t)nt * KT) * 512 + (size_t)lane * 16; // 512 elems per kt

  v8f acc = {};
#pragma unroll
  for (int kt = 0; kt < KT; ++kt) {
    v8bf alo = *(const v8bf*)(arow + kt * 32 + khalf);
    v8bf ahi = *(const v8bf*)(arow + kt * 32 + khalf + 16);
    const bf16* bp = bbase + (size_t)kt * 512;
    v8bf blo = *(const v8bf*)bp;
    v8bf bhi = *(const v8bf*)(bp + 8);
    acc = __builtin_amdgcn_wmma_f32_16x16x32_bf16(false, mk16(alo, ahi),
                                                  false, mk16(blo, bhi),
                                                  (short)0, acc, false, false);
  }

  const int n     = nt * 16 + rc;
  const int mbase = mt * 16 + ((lane >> 4) << 3);
  const float bv  = bias[n];
#pragma unroll
  for (int r = 0; r < 8; ++r)
    Craw[(size_t)(mbase + r) * N + n] = acc[r] + bv;
}

// ---------------------------------------------------------------------------
// Input-projection GEMM (throughput path):
//   Craw[M,N] = x[M,K](fp32) @ Wi[K,N] + bi,  M=32768, K=256, N=2048
// fp32 A tile staged->bf16 in LDS (reused by all waves); B frags direct from
// pre-swizzled Wif (L2-resident). Block = 64(M) x 128(N); each wave owns
// 4 N-tiles -> 4 WMMAs per A fragment per K chunk.
// ---------------------------------------------------------------------------
__global__ __launch_bounds__(256)
void wmma_gemm_xi(const float* __restrict__ A, const bf16* __restrict__ Bf,
                  const float* __restrict__ bias, float* __restrict__ Craw,
                  int M, int N, int K) {
  constexpr int AS = 40;  // padded LDS row stride (80B -> 16B-aligned frags)
  __shared__ __align__(16) bf16 Asm[64 * AS];

  const int tid  = threadIdx.x;
  const int lane = tid & 31;
  const int wave = tid >> 5;
  const int mt   = wave & 3;    // M sub-tile
  const int nth  = wave >> 2;   // N half: owns tiles nth*4 .. nth*4+3
  const int m0   = blockIdx.y * 64;
  const int nt0  = blockIdx.x * 8;  // 8 N-tiles (128 cols) per block
  const int KT   = K >> 5;

  const int rc    = lane & 15;
  const int khalf = (lane >> 4) << 3;

  v8f acc[4] = {};

  for (int kt = 0; kt < KT; ++kt) {
    // stage A tile 64x32 (coalesced over k), convert fp32 -> bf16
#pragma unroll
    for (int j = 0; j < 8; ++j) {
      int i  = tid + j * 256;
      int am = i >> 5;
      int ak = i & 31;
      Asm[am * AS + ak] = f2bf(A[(size_t)(m0 + am) * K + (size_t)(kt * 32 + ak)]);
    }
    __syncthreads();

    const bf16* ap = &Asm[(mt * 16 + rc) * AS + khalf];
    v16bf af = mk16(*(const v8bf*)ap, *(const v8bf*)(ap + 16));
#pragma unroll
    for (int q = 0; q < 4; ++q) {
      int nt = nt0 + nth * 4 + q;
      const bf16* bp = Bf + ((size_t)nt * KT + kt) * 512 + (size_t)lane * 16;
      v16bf bfrag = mk16(*(const v8bf*)bp, *(const v8bf*)(bp + 8));
      acc[q] = __builtin_amdgcn_wmma_f32_16x16x32_bf16(false, af, false, bfrag,
                                                       (short)0, acc[q], false, false);
    }
    __syncthreads();
  }

  const int mbase = m0 + mt * 16 + ((lane >> 4) << 3);
#pragma unroll
  for (int q = 0; q < 4; ++q) {
    int n = (nt0 + nth * 4 + q) * 16 + rc;
    float bv = bias[n];
#pragma unroll
    for (int r = 0; r < 8; ++r)
      Craw[(size_t)(mbase + r) * N + n] = acc[q][r] + bv;
  }
}

// ---------------------------------------------------------------------------
// In-place LayerNorm over rows of width W with affine (g,b). 1 block per row.
// ---------------------------------------------------------------------------
__global__ __launch_bounds__(256)
void ln_rows(float* __restrict__ y, const float* __restrict__ g,
             const float* __restrict__ b, int W) {
  float* p = y + (size_t)blockIdx.x * (size_t)W;
  const int tid = threadIdx.x;
  __shared__ float rs[256], rss[256];
  float s = 0.f, ss = 0.f;
  for (int i = tid; i < W; i += 256) { float v = p[i]; s += v; ss += v * v; }
  rs[tid] = s; rss[tid] = ss; __syncthreads();
  for (int off = 128; off > 0; off >>= 1) {
    if (tid < off) { rs[tid] += rs[tid + off]; rss[tid] += rss[tid + off]; }
    __syncthreads();
  }
  float mean = rs[0] / (float)W;
  float var  = rss[0] / (float)W - mean * mean;
  float rstd = rsqrtf(var + LN_EPS);
  for (int i = tid; i < W; i += 256)
    p[i] = (p[i] - mean) * rstd * g[i] + b[i];
}

// ---------------------------------------------------------------------------
// Fused per-timestep cell: LN(pre_raw) + xi_t -> gates -> c update -> LN(c)
// -> h. One block per batch row b. H=512, 4H=2048, 256 threads.
// ---------------------------------------------------------------------------
__global__ __launch_bounds__(256)
void lstm_step(const float* __restrict__ pre_raw,   // [B,4H] = h@Wh + bh
               const float* __restrict__ xi,        // [B,T,4H] (already LN'd)
               const float* __restrict__ g2, const float* __restrict__ b2,
               const float* __restrict__ g3, const float* __restrict__ b3,
               float* __restrict__ c_state,         // [B,H]
               bf16*  __restrict__ h_bf,            // [B,H] next-step A operand
               float* __restrict__ O,               // [B,T,H]
               int t, int T, int H) {
  const int b   = blockIdx.x;
  const int tid = threadIdx.x;
  const int W   = 4 * H;  // 2048
  __shared__ float raw[2048];
  __shared__ float rs[256], rss[256];

  const float* pr = pre_raw + (size_t)b * (size_t)W;
  float s = 0.f, ss = 0.f;
  for (int i = tid; i < W; i += 256) { float v = pr[i]; raw[i] = v; s += v; ss += v * v; }
  rs[tid] = s; rss[tid] = ss; __syncthreads();
  for (int off = 128; off > 0; off >>= 1) {
    if (tid < off) { rs[tid] += rs[tid + off]; rss[tid] += rss[tid + off]; }
    __syncthreads();
  }
  const float mean = rs[0] / (float)W;
  const float var  = rss[0] / (float)W - mean * mean;
  const float rstd = rsqrtf(var + LN_EPS);
  __syncthreads();  // rs/rss reused below

  const float* xr = xi + ((size_t)b * (size_t)T + (size_t)t) * (size_t)W;
  float cr[2], og[2];
  float s2 = 0.f, ss2 = 0.f;
#pragma unroll
  for (int kk = 0; kk < 2; ++kk) {
    int j = tid + kk * 256;
    float pf = (raw[j        ] - mean) * rstd * g2[j        ] + b2[j        ] + xr[j        ];
    float pi = (raw[j +     H] - mean) * rstd * g2[j +     H] + b2[j +     H] + xr[j +     H];
    float po = (raw[j + 2 * H] - mean) * rstd * g2[j + 2 * H] + b2[j + 2 * H] + xr[j + 2 * H];
    float pg = (raw[j + 3 * H] - mean) * rstd * g2[j + 3 * H] + b2[j + 3 * H] + xr[j + 3 * H];
    float f  = 1.0f / (1.0f + __expf(-pf));
    float ig = 1.0f / (1.0f + __expf(-pi));
    float o  = 1.0f / (1.0f + __expf(-po));
    float g  = tanhf(pg);
    float c  = c_state[(size_t)b * H + j];
    float cn = c * f + ig * g;
    cr[kk] = cn; og[kk] = o;
    s2 += cn; ss2 += cn * cn;
  }
  rs[tid] = s2; rss[tid] = ss2; __syncthreads();
  for (int off = 128; off > 0; off >>= 1) {
    if (tid < off) { rs[tid] += rs[tid + off]; rss[tid] += rss[tid + off]; }
    __syncthreads();
  }
  const float m2 = rs[0] / (float)H;
  const float v2 = rss[0] / (float)H - m2 * m2;
  const float r2 = rsqrtf(v2 + LN_EPS);
#pragma unroll
  for (int kk = 0; kk < 2; ++kk) {
    int j = tid + kk * 256;
    float cn = (cr[kk] - m2) * r2 * g3[j] + b3[j];
    c_state[(size_t)b * H + j] = cn;
    float h = og[kk] * tanhf(cn);
    O[((size_t)b * (size_t)T + (size_t)t) * (size_t)H + j] = h;
    h_bf[(size_t)b * H + j] = f2bf(h);
  }
}

// copy final h (= O[:,T-1,:]) and c into output tail
__global__ __launch_bounds__(256)
void finalize_hc(const float* __restrict__ O, const float* __restrict__ c_state,
                 float* __restrict__ hout, float* __restrict__ cout,
                 int B, int T, int H) {
  int i = blockIdx.x * 256 + threadIdx.x;
  if (i < B * H) {
    int b = i / H, j = i % H;
    hout[i] = O[((size_t)b * (size_t)T + (size_t)(T - 1)) * (size_t)H + j];
    cout[i] = c_state[i];
  }
}

// ---------------------------------------------------------------------------
// Launch
// ---------------------------------------------------------------------------
extern "C" void kernel_launch(void* const* d_in, const int* in_sizes, int n_in,
                              void* d_out, int out_size, void* d_ws, size_t ws_size,
                              hipStream_t stream) {
  (void)in_sizes; (void)n_in; (void)out_size; (void)ws_size;
  const int B = 64, T = 512, D = 256, H = 512;
  const int FH = 4 * H;           // 2048
  const int MT = B * T;           // 32768

  const float* x  = (const float*)d_in[0];
  const float* Wi = (const float*)d_in[1];
  const float* bi = (const float*)d_in[2];
  const float* Wh = (const float*)d_in[3];
  const float* bh = (const float*)d_in[4];
  const float* g1 = (const float*)d_in[5];
  const float* b1 = (const float*)d_in[6];
  const float* g2 = (const float*)d_in[7];
  const float* b2 = (const float*)d_in[8];
  const float* g3 = (const float*)d_in[9];
  const float* b3 = (const float*)d_in[10];

  float* O    = (float*)d_out;            // [B,T,H]
  float* hout = O + (size_t)B * T * H;    // [B,H]
  float* cout = hout + (size_t)B * H;     // [B,H]

  // workspace partition (256B aligned)
  uintptr_t p = (uintptr_t)d_ws;
  auto alloc = [&](size_t bytes) -> void* {
    p = (p + 255) & ~(uintptr_t)255;
    void* r = (void*)p; p += bytes; return r;
  };
  bf16*  Wif     = (bf16*)alloc((size_t)D * FH * sizeof(bf16));   // swizzled frags
  bf16*  Whf     = (bf16*)alloc((size_t)H * FH * sizeof(bf16));   // swizzled frags
  float* xi      = (float*)alloc((size_t)MT * FH * sizeof(float)); // 256 MB
  float* pre     = (float*)alloc((size_t)B * FH * sizeof(float));
  float* c_state = (float*)alloc((size_t)B * H * sizeof(float));
  bf16*  h_bf    = (bf16*)alloc((size_t)B * H * sizeof(bf16));

  // 1) weights -> bf16 B-fragment swizzle (one-time; L2-resident thereafter)
  swizzle_wf<<<(D * FH + 255) / 256, 256, 0, stream>>>(Wi, Wif, D, FH);
  swizzle_wf<<<(H * FH + 255) / 256, 256, 0, stream>>>(Wh, Whf, H, FH);
  // 2) h0 = 0, c0 = 0
  init_state<<<(B * H + 255) / 256, 256, 0, stream>>>(c_state, h_bf, B * H);

  // 3) xi_raw = x @ Wi + bi   (32768 x 2048, K=256)
  wmma_gemm_xi<<<dim3(FH / 128, MT / 64), 256, 0, stream>>>(x, Wif, bi, xi, MT, FH, D);
  // 4) xi = LN(xi_raw) * g1 + b1 (in place)
  ln_rows<<<MT, 256, 0, stream>>>(xi, g1, b1, FH);

  // 5) recurrent scan: register-direct WMMA GEMM + fused cell, 2 launches/step
  for (int t = 0; t < T; ++t) {
    wmma_gemm_step<<<64, 256, 0, stream>>>(h_bf, Whf, bh, pre);
    lstm_step<<<B, 256, 0, stream>>>(pre, xi, g2, b2, g3, b3,
                                     c_state, h_bf, O, t, T, H);
  }

  // 6) emit final h, c
  finalize_hc<<<(B * H + 255) / 256, 256, 0, stream>>>(O, c_state, hout, cout, B, T, H);
}
